// ConvAttention_86466281603481
// MI455X (gfx1250) — matrix operand order repaired and checked by reading, not compiled, not run on hardware
//
#include <hip/hip_runtime.h>
#include <hip/hip_fp16.h>

#define B_  4
#define S_  2048
#define C_  128
#define H_  8
#define DK_ 64
#define DV_ 64
#define KC_ 3

#define LDP 2052                      // LDS row stride (floats) for logit strip
// logits (16*LDP) + red(256) + rowmax(16) + rowscl(16) + partial(4*256)
#define ATTN_LDS ((16 * LDP + 256 + 16 + 16 + 1024) * 4)
#define MLP_LDS  (512 * 65 * 4)

typedef _Float16 v16h __attribute__((ext_vector_type(16)));
typedef _Float16 v8h  __attribute__((ext_vector_type(8)));
typedef float    v8f  __attribute__((ext_vector_type(8)));

static __device__ inline v8f wmma_f16(v16h a, v16h b, v8f c) {
  return __builtin_amdgcn_wmma_f32_16x16x32_f16(false, a, false, b, (short)0, c,
                                                false, false);
}

// A fragment (16x32, f16, row-major src with leading dim ld, chunk at col c0).
// lane<16: row=lane, K = c0+[0..7] and c0+[16..23]
// lane>=16: row=lane-16, K = c0+[8..15] and c0+[24..31]
static __device__ inline v16h load_frag_a(const _Float16* __restrict__ src,
                                          int ld, int c0, int lane) {
  const int r = lane & 15;
  const _Float16* base = src + r * ld + c0 + ((lane & 16) ? 8 : 0);
  v8h lo = *(const v8h*)(base);
  v8h hi = *(const v8h*)(base + 16);
  return __builtin_shufflevector(lo, hi, 0, 1, 2, 3, 4, 5, 6, 7,
                                 8, 9, 10, 11, 12, 13, 14, 15);
}

// B fragment (32x16, f16). src rows indexed by the N dimension (i.e. K-matrix
// rows, or V^T rows), contiguous along the WMMA K dimension; chunk at col c0.
// lane<16: col=lane, K = c0+[0..7] and c0+[16..23]
// lane>=16: col=lane-16, K = c0+[8..15] and c0+[24..31]
static __device__ inline v16h load_frag_b(const _Float16* __restrict__ src,
                                          int ld, int c0, int lane) {
  const int r = lane & 15;
  const _Float16* base = src + r * ld + c0 + ((lane & 16) ? 8 : 0);
  v8h lo = *(const v8h*)(base);
  v8h hi = *(const v8h*)(base + 16);
  return __builtin_shufflevector(lo, hi, 0, 1, 2, 3, 4, 5, 6, 7,
                                 8, 9, 10, 11, 12, 13, 14, 15);
}

// A fragment from f32 probabilities in LDS (convert to f16 on read).
static __device__ inline v16h load_frag_p(const float* __restrict__ lds,
                                          int ldl, int c0, int lane) {
  const int r = lane & 15;
  const float* base = lds + r * ldl + c0 + ((lane & 16) ? 8 : 0);
  v16h a;
#pragma unroll
  for (int i = 0; i < 8; ++i) {
    a[i]     = (_Float16)base[i];
    a[i + 8] = (_Float16)base[i + 16];
  }
  return a;
}

// ---------------------------------------------------------------------------
// Kernel 1: Q/K causal conv (K=3) + V = relu(x@wv^T+bv).
// Q,K stored f16 (b,h,s,dk); V stored f16 transposed (b,h,dv,s).
// ---------------------------------------------------------------------------
__global__ __launch_bounds__(256) void qkv_kernel(
    const float* __restrict__ x,
    const float* __restrict__ wq, const float* __restrict__ bq,
    const float* __restrict__ wk, const float* __restrict__ bk,
    const float* __restrict__ wv, const float* __restrict__ bv,
    _Float16* __restrict__ qh, _Float16* __restrict__ kh,
    _Float16* __restrict__ vT) {
  __shared__ float xs[34][C_];  // rows map to s0-2 .. s0+31
  const int tid = threadIdx.x;
  const int nbs = S_ / 32;
  const int b = blockIdx.x / nbs;
  const int s0 = (blockIdx.x % nbs) * 32;

  for (int idx = tid; idx < 34 * C_; idx += 256) {
    int j = idx >> 7, c = idx & (C_ - 1);
    int s = s0 - 2 + j;
    xs[j][c] = (s >= 0) ? x[((size_t)b * S_ + s) * C_ + c] : 0.0f;
  }
  __syncthreads();

  for (int it = 0; it < 6; ++it) {
    const int oid = it * 256 + tid;   // 0..1535
    const int mat = oid >> 9;         // 0=q, 1=k, 2=v
    const int o = oid & 511;
    const int h = o >> 6, d = o & 63;
    float acc[32];
    if (mat < 2) {
      const float* w = (mat ? wk : wq) + (size_t)o * C_ * KC_;
      const float bias = (mat ? bk : bq)[o];
#pragma unroll
      for (int p = 0; p < 32; ++p) acc[p] = bias;
      for (int c = 0; c < C_; ++c) {
        const float w0 = w[c * 3 + 0], w1 = w[c * 3 + 1], w2 = w[c * 3 + 2];
#pragma unroll
        for (int p = 0; p < 32; ++p)
          acc[p] += w0 * xs[p][c] + w1 * xs[p + 1][c] + w2 * xs[p + 2][c];
      }
      _Float16* dst = mat ? kh : qh;
#pragma unroll
      for (int p = 0; p < 32; ++p)
        dst[(((size_t)(b * H_ + h)) * S_ + (s0 + p)) * DK_ + d] =
            (_Float16)acc[p];
    } else {
      const float* w = wv + (size_t)o * C_;
      const float bias = bv[o];
#pragma unroll
      for (int p = 0; p < 32; ++p) acc[p] = bias;
      for (int c = 0; c < C_; ++c) {
        const float wc = w[c];
#pragma unroll
        for (int p = 0; p < 32; ++p) acc[p] += wc * xs[p + 2][c];
      }
#pragma unroll
      for (int p = 0; p < 32; ++p)
        vT[(((size_t)(b * H_ + h)) * DV_ + d) * S_ + (s0 + p)] =
            (_Float16)fmaxf(acc[p], 0.0f);
    }
  }
}

// ---------------------------------------------------------------------------
// Kernel 2: per (b,h,16-row q tile): QK^T (WMMA) -> LDS logits -> softmax
// in LDS -> stream score strip (f32, causal zeros) -> P@V (WMMA) -> out2.
// ---------------------------------------------------------------------------
__global__ __launch_bounds__(256) void attn_kernel(
    const _Float16* __restrict__ qh, const _Float16* __restrict__ kh,
    const _Float16* __restrict__ vT, const int* __restrict__ length,
    float* __restrict__ score, float* __restrict__ out2) {
  extern __shared__ float smem[];
  float* logits  = smem;                    // 16 * LDP
  float* red     = smem + 16 * LDP;         // 256
  float* rowmax  = red + 256;               // 16
  float* rowscl  = rowmax + 16;             // 16
  float* partial = rowscl + 16;             // 4 * 256 (P@V cross-wave reduce)

  const int tid = threadIdx.x;
  const int lane = tid & 31;
  // force wave id into an SGPR: scalar loops/branches, EXEC untouched at WMMA
  const int wave = __builtin_amdgcn_readfirstlane(tid >> 5);
  const int qt = blockIdx.x;
  const int q0 = qt * 16;
  const int h = blockIdx.y;
  const int b = blockIdx.z;
  const int len = length[b];

  const size_t headoff = ((size_t)(b * H_ + h)) * S_;
  const _Float16* qbase = qh + (headoff + q0) * DK_;
  const v16h a0 = load_frag_a(qbase, DK_, 0, lane);
  const v16h a1 = load_frag_a(qbase, DK_, 32, lane);

  // ---- QK^T into LDS (causal tiles only), scaled + masked --------------
  for (int jt = wave; jt <= qt; jt += 8) {
    const _Float16* kbase = kh + (headoff + jt * 16) * DK_;
    const v16h b0 = load_frag_b(kbase, DK_, 0, lane);
    const v16h b1 = load_frag_b(kbase, DK_, 32, lane);
    v8f c = {};
    c = wmma_f16(a0, b0, c);
    c = wmma_f16(a1, b1, c);
    const int n = lane & 15;
    const int mofs = (lane & 16) ? 8 : 0;
#pragma unroll
    for (int r = 0; r < 8; ++r) {
      const int m = r + mofs;
      const int j = jt * 16 + n;
      float v = c[r] * 0.125f;  // 1/sqrt(64)
      if (j > q0 + m) v = -1.0e30f;
      logits[m * LDP + j] = v;
    }
  }
  __syncthreads();

  // ---- softmax over LDS strip ------------------------------------------
  const int ncols = (qt + 1) * 16;
  const int row = tid & 15;
  const int slice = tid >> 4;  // 16 slices per row

  float mx = -3.0e38f;
  for (int j = slice; j < ncols; j += 16)
    mx = fmaxf(mx, logits[row * LDP + j]);
  red[row * 16 + slice] = mx;
  __syncthreads();
  if (slice == 0) {
    float m2 = red[row * 16];
#pragma unroll
    for (int t = 1; t < 16; ++t) m2 = fmaxf(m2, red[row * 16 + t]);
    rowmax[row] = m2;
  }
  __syncthreads();
  const float rm = rowmax[row];
  float sum = 0.0f;
  for (int j = slice; j < ncols; j += 16) {
    const float e = __expf(logits[row * LDP + j] - rm);
    logits[row * LDP + j] = e;
    sum += e;
  }
  red[row * 16 + slice] = sum;
  __syncthreads();
  if (slice == 0) {
    float s2 = 0.0f;
#pragma unroll
    for (int t = 0; t < 16; ++t) s2 += red[row * 16 + t];
    rowscl[row] = (q0 + row < len) ? (1.0f / s2) : 0.0f;  // pad-mask rows
  }
  __syncthreads();
  const float inv = rowscl[row];
  for (int j = slice; j < ncols; j += 16) logits[row * LDP + j] *= inv;
  // pad 16 zero columns when #tiles is odd so K-chunks of 32 are clean
  if ((qt & 1) == 0) logits[(tid >> 4) * LDP + ncols + (tid & 15)] = 0.0f;
  __syncthreads();

  // ---- stream the full 16 x S score strip (f32, coalesced float4) -------
  {
    float* srow = score + (headoff + q0) * S_;
    for (int idx = tid; idx < 16 * (S_ / 4); idx += 256) {
      const int r = idx >> 9;
      const int j = (idx & (S_ / 4 - 1)) * 4;
      float4 v;
      if (j < ncols)
        v = *reinterpret_cast<const float4*>(&logits[r * LDP + j]);
      else
        v.x = v.y = v.z = v.w = 0.0f;
      *reinterpret_cast<float4*>(&srow[(size_t)r * S_ + j]) = v;
    }
  }

  // ---- out1 = P @ V : 8 waves = 4 d-tiles x 2 K-chunk parities ----------
  const int dtile = wave & 3;
  const int khalf = wave >> 2;        // K-chunk parity (interleaved split)
  const int d0 = dtile * 16;
  const _Float16* vbase = vT + (((size_t)(b * H_ + h)) * DV_ + d0) * S_;
  v8f c = {};
  for (int k0 = khalf * 32; k0 < ncols; k0 += 64) {
    const v16h a = load_frag_p(logits, LDP, k0, lane);
    const v16h bb = load_frag_b(vbase, S_, k0, lane);
    c = wmma_f16(a, bb, c);
  }
  if (khalf == 1) {
#pragma unroll
    for (int r = 0; r < 8; ++r) partial[dtile * 256 + r * 32 + lane] = c[r];
  }
  __syncthreads();
  if (khalf == 0) {
    const int n = lane & 15;
    const int mofs = (lane & 16) ? 8 : 0;
#pragma unroll
    for (int r = 0; r < 8; ++r) {
      const int m = r + mofs;
      const float v = c[r] + partial[dtile * 256 + r * 32 + lane];
      out2[((size_t)(b * S_ + q0 + m)) * (H_ * DV_) + h * DV_ + d0 + n] = v;
    }
  }
}

// ---------------------------------------------------------------------------
// Kernel 3: out3 = relu(out2 @ w1^T + b1) @ w2^T + b2
// ---------------------------------------------------------------------------
__global__ __launch_bounds__(256) void mlp_kernel(
    const float* __restrict__ out2, const float* __restrict__ w1,
    const float* __restrict__ b1, const float* __restrict__ w2,
    const float* __restrict__ b2, float* __restrict__ out3) {
  extern __shared__ float w1T[];  // [512][65] padded transpose of w1(64,512)
  const int tid = threadIdx.x;
  for (int idx = tid; idx < 64 * 512; idx += 256) {
    const int d = idx >> 9;
    const int e = idx & 511;
    w1T[e * 65 + d] = w1[idx];  // coalesced global read
  }
  __syncthreads();

  const int pos = blockIdx.x * 64 + (tid >> 2);  // (b*S+s)
  const int qq = tid & 3;                        // 4 lanes per position
  const float* xrow = out2 + (size_t)pos * 512;
  float acc[16];
#pragma unroll
  for (int dl = 0; dl < 16; ++dl) acc[dl] = b1[qq * 16 + dl];
  for (int e = 0; e < 512; ++e) {
    const float xv = xrow[e];
    const float* wr = &w1T[e * 65 + qq * 16];
#pragma unroll
    for (int dl = 0; dl < 16; ++dl) acc[dl] += wr[dl] * xv;
  }
  float part = 0.0f;
#pragma unroll
  for (int dl = 0; dl < 16; ++dl)
    part += w2[qq * 16 + dl] * fmaxf(acc[dl], 0.0f);
  part += __shfl_xor(part, 1, 32);
  part += __shfl_xor(part, 2, 32);
  if (qq == 0) out3[pos] = part + b2[0];
}

// ---------------------------------------------------------------------------
extern "C" void kernel_launch(void* const* d_in, const int* in_sizes, int n_in,
                              void* d_out, int out_size, void* d_ws,
                              size_t ws_size, hipStream_t stream) {
  (void)in_sizes; (void)n_in; (void)out_size; (void)ws_size;

  const float* x   = (const float*)d_in[0];
  const int* length = (const int*)d_in[1];
  const float* wq  = (const float*)d_in[2];
  const float* bq  = (const float*)d_in[3];
  const float* wk  = (const float*)d_in[4];
  const float* bk  = (const float*)d_in[5];
  const float* wv  = (const float*)d_in[6];
  const float* bv  = (const float*)d_in[7];
  const float* w1  = (const float*)d_in[8];
  const float* b1  = (const float*)d_in[9];
  const float* w2  = (const float*)d_in[10];
  const float* b2  = (const float*)d_in[11];

  float* out3 = (float*)d_out;                       // (B,S,1)
  float* score = out3 + (size_t)B_ * S_;             // (B,H,S,S)

  char* ws = (char*)d_ws;
  const size_t qkv_bytes = (size_t)B_ * H_ * S_ * DK_ * sizeof(_Float16);
  _Float16* qh = (_Float16*)(ws);
  _Float16* kh = (_Float16*)(ws + qkv_bytes);
  _Float16* vT = (_Float16*)(ws + 2 * qkv_bytes);
  float* out2  = (float*)(ws + 3 * qkv_bytes);       // (B,S,H*DV) f32

  hipFuncSetAttribute(reinterpret_cast<const void*>(attn_kernel),
                      hipFuncAttributeMaxDynamicSharedMemorySize, ATTN_LDS);
  hipFuncSetAttribute(reinterpret_cast<const void*>(mlp_kernel),
                      hipFuncAttributeMaxDynamicSharedMemorySize, MLP_LDS);

  qkv_kernel<<<dim3(B_ * S_ / 32), dim3(256), 0, stream>>>(
      x, wq, bq, wk, bk, wv, bv, qh, kh, vT);
  attn_kernel<<<dim3(S_ / 16, H_, B_), dim3(256), ATTN_LDS, stream>>>(
      qh, kh, vT, length, score, out2);
  mlp_kernel<<<dim3(B_ * S_ / 64), dim3(256), MLP_LDS, stream>>>(
      out2, w1, b1, w2, b2, out3);
}
